// MultiHeadAttention_86234353369683
// MI455X (gfx1250) — matrix operand order repaired and checked by reading, not compiled, not run on hardware
//
#include <hip/hip_runtime.h>
#include <hip/hip_bf16.h>

// ---------------------------------------------------------------------------
// Types for CDNA5 WMMA (wave32): v_wmma_f32_16x16x32_bf16  + TDM descriptors
// ---------------------------------------------------------------------------
typedef __attribute__((ext_vector_type(16))) __bf16 v16bf;
typedef __attribute__((ext_vector_type(8)))  float  v8f;
typedef __attribute__((ext_vector_type(4)))  unsigned int v4u;
typedef __attribute__((ext_vector_type(8)))  int    v8i;
typedef __attribute__((ext_vector_type(4)))  int    v4i;

union FragB16 { v16bf v; uint4 q[2]; };
union U8x16  { uint4 q; unsigned short s[8]; };

static __device__ inline unsigned short f2bf(float f) {
    union { float f; unsigned int u; } v; v.f = f;
    unsigned int r = v.u + 0x7FFFu + ((v.u >> 16) & 1u);   // round-to-nearest-even
    return (unsigned short)(r >> 16);
}

static __device__ inline v8f wmma_bf16(v16bf a, v16bf b, v8f c) {
    return __builtin_amdgcn_wmma_f32_16x16x32_bf16(
        false, a, false, b, (short)0, c, false, false);
}

// A-fragment (16x32 bf16, MxK) from LDS laid out row-major [m][k], stride in elems.
// ISA 7.12.2: lanes 0-15 row=lane hold K=0..7 (v0-3) & K=16..23 (v4-7);
//             lanes 16-31 row=lane-16 hold K=8..15 & K=24..31.
static __device__ inline v16bf load_a_frag(const unsigned short* base, int stride,
                                           int koff, int lane) {
    int row = lane & 15, half = lane >> 4;
    const unsigned short* p = base + row * stride + koff;
    FragB16 f;
    f.q[0] = *(const uint4*)(p + half * 8);
    f.q[1] = *(const uint4*)(p + 16 + half * 8);
    return f.v;
}

// B-fragment (32x16 bf16, KxN) from LDS laid out [n][k] (k contiguous per column).
// lanes 0-15 col=lane hold K=0..7 & K=16..23; lanes 16-31 col=lane-16 hold K=8..15 & K=24..31.
static __device__ inline v16bf load_b_frag(const unsigned short* base, int stride,
                                           int koff, int lane) {
    int n = lane & 15, half = lane >> 4;
    const unsigned short* p = base + n * stride + koff;
    FragB16 f;
    f.q[0] = *(const uint4*)(p + half * 8);
    f.q[1] = *(const uint4*)(p + 16 + half * 8);
    return f.v;
}

// fp32 -> 4x bf16, or pass-through bf16 raw bits
static __device__ inline ushort4 ld_bf4(const float* p) {
    float4 f = *(const float4*)p;
    ushort4 r; r.x = f2bf(f.x); r.y = f2bf(f.y); r.z = f2bf(f.z); r.w = f2bf(f.w);
    return r;
}
static __device__ inline ushort4 ld_bf4(const unsigned short* p) {
    return *(const ushort4*)p;
}
static __device__ inline void st_out(float* p, float v) { *p = v; }
static __device__ inline void st_out(unsigned short* p, float v) { *p = f2bf(v); }

// ---------------------------------------------------------------------------
// GEMM + bias:  C[M,N] = A[M,K] @ W[K,N] + bias[N]
// A: fp32 or bf16-bits; W,bias: fp32; C: fp32 or bf16-bits.
// 256 threads = 8 waves (4x2 wave grid), 128x128x64 tiles, bf16 WMMA,
// software-pipelined: next tile staged in VGPRs while WMMAs run on LDS tile.
// W-tile transposed into LDS with packed ds_store_b64 (4 consecutive k per n).
// ---------------------------------------------------------------------------
template <typename AT, typename OT>
__global__ __launch_bounds__(256)
void gemm_bias_kernel(const AT* __restrict__ A, const float* __restrict__ W,
                      const float* __restrict__ bias, OT* __restrict__ C,
                      int M, int N, int K)
{
    constexpr int BK = 64, LDK = 72;
    __shared__ __align__(16) unsigned short As[128][LDK];   // [m][k]
    __shared__ __align__(16) unsigned short Bs[128][LDK];   // [n][k] (transposed)

    const int tid  = threadIdx.x;
    const int lane = tid & 31, wave = tid >> 5;
    const int wm = wave >> 1;            // 0..3  (M direction, 32 rows each)
    const int wn = wave & 1;             // 0..1  (N direction, 64 cols each)
    const int m0 = blockIdx.y * 128;
    const int n0 = blockIdx.x * 128;

    // staging maps
    const int ar = tid >> 1, ac = (tid & 1) * 32;   // A tile: 2 thr/row x 32 elems
    const int kq = tid >> 4;                        // W tile: k rows 4kq..4kq+3
    const int nb = (tid & 15) * 8;                  //         n cols nb..nb+7

    ushort4 avs[8];
    unsigned short wstage[4][8];
    auto load_stage = [&](int k0) {
#pragma unroll
        for (int p = 0; p < 8; ++p)
            avs[p] = ld_bf4(A + (size_t)(m0 + ar) * K + k0 + ac + p * 4);
#pragma unroll
        for (int rr = 0; rr < 4; ++rr) {
            const float* wp = &W[(size_t)(k0 + 4 * kq + rr) * N + n0 + nb];
            float4 f0 = *(const float4*)(wp);
            float4 f1 = *(const float4*)(wp + 4);
            wstage[rr][0] = f2bf(f0.x); wstage[rr][1] = f2bf(f0.y);
            wstage[rr][2] = f2bf(f0.z); wstage[rr][3] = f2bf(f0.w);
            wstage[rr][4] = f2bf(f1.x); wstage[rr][5] = f2bf(f1.y);
            wstage[rr][6] = f2bf(f1.z); wstage[rr][7] = f2bf(f1.w);
        }
    };
    auto store_stage = [&]() {
#pragma unroll
        for (int p = 0; p < 8; ++p)
            *(ushort4*)&As[ar][ac + p * 4] = avs[p];
#pragma unroll
        for (int n = 0; n < 8; ++n) {
            ushort4 u;
            u.x = wstage[0][n]; u.y = wstage[1][n];
            u.z = wstage[2][n]; u.w = wstage[3][n];
            *(ushort4*)&Bs[nb + n][4 * kq] = u;    // 4 consecutive k, one b64 store
        }
    };

    v8f zero = {0.f, 0.f, 0.f, 0.f, 0.f, 0.f, 0.f, 0.f};
    v8f acc[2][4];
#pragma unroll
    for (int i = 0; i < 2; ++i)
#pragma unroll
        for (int j = 0; j < 4; ++j) acc[i][j] = zero;

    load_stage(0);
    const int nIter = K / BK;
    for (int it = 0; it < nIter; ++it) {
        store_stage();
        __syncthreads();
        if (it + 1 < nIter) load_stage((it + 1) * BK);   // overlap with WMMAs below
#pragma unroll
        for (int c = 0; c < 2; ++c) {
            v16bf af[2], bfr[4];
#pragma unroll
            for (int i = 0; i < 2; ++i)
                af[i] = load_a_frag(&As[wm * 32 + i * 16][0], LDK, c * 32, lane);
#pragma unroll
            for (int j = 0; j < 4; ++j)
                bfr[j] = load_b_frag(&Bs[wn * 64 + j * 16][0], LDK, c * 32, lane);
#pragma unroll
            for (int i = 0; i < 2; ++i)
#pragma unroll
                for (int j = 0; j < 4; ++j)
                    acc[i][j] = wmma_bf16(af[i], bfr[j], acc[i][j]);
        }
        __syncthreads();
    }

    // --- epilogue: bias add + store (D layout: row = r + 8*(lane>>4), col = lane&15) ---
    const int half = lane >> 4, nl = lane & 15;
#pragma unroll
    for (int i = 0; i < 2; ++i)
#pragma unroll
        for (int j = 0; j < 4; ++j) {
            const int col = n0 + wn * 64 + j * 16 + nl;
            const float bv = bias[col];
#pragma unroll
            for (int r = 0; r < 8; ++r) {
                const int row = m0 + wm * 32 + i * 16 + r + 8 * half;
                st_out(C + (size_t)row * N + col, acc[i][j][r] + bv);
            }
        }
}

// ---------------------------------------------------------------------------
// Flash attention: one WG per (b, h, 128 q rows), 8 waves x 16 q-rows.
// q/k/v/o are bf16 bits, layout (B*S, 1024), head h occupies cols [h*64, h*64+64).
// KV streamed in 64-key tiles, DOUBLE-BUFFERED: the Tensor Data Mover fetches
// the next K tile (tensor_load_to_lds -> padded LDS rows, TENSORcnt) and the
// waves stage the next V tile in VGPRs while the current tile's 16 WMMAs and
// softmax run; one barrier per tile. Online softmax in exp2 domain.
// ---------------------------------------------------------------------------
__global__ __launch_bounds__(256)
void flash_attn_kernel(const unsigned short* __restrict__ Qb,
                       const unsigned short* __restrict__ Kb,
                       const unsigned short* __restrict__ Vb,
                       unsigned short* __restrict__ Ob)
{
    constexpr int S = 2048, D = 1024, HD = 64;
    constexpr float cs = 0.18033688011112042f;   // (1/sqrt(64)) * log2(e)
    constexpr unsigned int KBYTES = 64 * 72 * 2; // one K buffer: 9216 B

    __shared__ __align__(16) unsigned short Ks[2][64][72];   // [buf][kv][d] (TDM dest)
    __shared__ __align__(16) unsigned short Vt[2][64][72];   // [buf][d][e]  (transposed, permuted kv)
    __shared__ __align__(16) unsigned short Ps[8][16][72];   // per-wave P (16 x 64, permuted kv)

    const int tid  = threadIdx.x;
    const int lane = tid & 31, wave = tid >> 5;
    const int half = lane >> 4, nl = lane & 15;
    const int q0 = blockIdx.x * 128;
    const int h  = blockIdx.y;
    const int b  = blockIdx.z;
    const size_t baseRow = (size_t)b * S;
    const int hoff = h * HD;

    // resident Q fragments (16 rows x 64 dim -> two 16x32 A-frags)
    const int qrow = q0 + wave * 16 + nl;
    const unsigned short* qp = Qb + (baseRow + qrow) * D + hoff;
    v16bf qa[2];
#pragma unroll
    for (int c = 0; c < 2; ++c) {
        FragB16 f;
        f.q[0] = *(const uint4*)(qp + c * 32 + half * 8);
        f.q[1] = *(const uint4*)(qp + c * 32 + 16 + half * 8);
        qa[c] = f.v;
    }

    // TDM D# group1 (constant for all tiles): data_size=2B, pad 4 DWORDs every
    // 32 DWORDs (row 128B -> LDS stride 144B), tensor 64 x 2048, tile 64 x 64,
    // dim0 stride = D elements.
    const unsigned int g1w0 = (1u << 16) | (1u << 20) | (4u << 22) | (3u << 25);
    const v8i kg1 = { (int)g1w0,
                      (int)(64u << 16),          // tensor_dim0 = 64 (bits 79:48)
                      (int)((unsigned)S << 16),  // tensor_dim1   (bits 111:80)
                      (int)(64u << 16),          // tile_dim0 = 64 (bits 127:112)
                      64,                        // tile_dim1 = 64 (bits 143:128)
                      (int)D,                    // tensor_dim0_stride (bits 207:160)
                      0, 0 };
    const v4i gz  = { 0, 0, 0, 0 };
    const v8i gz8 = { 0, 0, 0, 0, 0, 0, 0, 0 };
    const unsigned int ldsK0 = (unsigned int)(size_t)&Ks[0][0][0];

    auto issue_k = [&](int kv0, int bufn) {    // wave 0 only
        unsigned long long ga =
            (unsigned long long)(size_t)(Kb + (baseRow + kv0) * D + hoff);
        v4u g0;
        g0.x = 1u;                                   // count=1 (valid user D#)
        g0.y = ldsK0 + (unsigned int)bufn * KBYTES;  // lds_addr
        g0.z = (unsigned int)ga;                     // global_addr[31:0]
        g0.w = (unsigned int)((ga >> 32) & 0x1FFFFFFu) | 0x80000000u; // [56:32] + type=2
        __builtin_amdgcn_tensor_load_to_lds(g0, kg1, gz, gz, gz8, 0);
    };

    // V staging: pair (kva, kva+16) packed per dword into permuted-kv layout
    const int vj = tid & 15, vg = (tid >> 4) & 1, vdc = (tid >> 5) * 8;
    U8x16 va, vb;
    auto load_v = [&](int kv0) {
        const int kva = kv0 + vj + 32 * vg;
        va.q = *(const uint4*)(Vb + (baseRow + kva)      * D + hoff + vdc);
        vb.q = *(const uint4*)(Vb + (baseRow + kva + 16) * D + hoff + vdc);
    };
    auto store_v = [&](int bufn) {
#pragma unroll
        for (int i = 0; i < 8; ++i) {
            unsigned int pk = (unsigned int)va.s[i] | ((unsigned int)vb.s[i] << 16);
            *(unsigned int*)&Vt[bufn][vdc + i][4 * vj + 2 * vg] = pk;
        }
    };

    v8f zero = {0.f, 0.f, 0.f, 0.f, 0.f, 0.f, 0.f, 0.f};
    v8f o[4];
#pragma unroll
    for (int f = 0; f < 4; ++f) o[f] = zero;
    float mrow[8], lrow[8];
#pragma unroll
    for (int r = 0; r < 8; ++r) { mrow[r] = -1e30f; lrow[r] = 0.f; }

    // prologue: fill buffer 0
    if (wave == 0) issue_k(0, 0);
    load_v(0);
    store_v(0);
    if (wave == 0) __builtin_amdgcn_s_wait_tensorcnt(0);
    __syncthreads();

    int buf = 0;
    for (int kv0 = 0; kv0 < S; kv0 += 64) {
        const int nxt = kv0 + 64;
        // kick next tile's DMA + V global loads before computing this tile
        if (nxt < S) {
            if (wave == 0) issue_k(nxt, buf ^ 1);
            load_v(nxt);
        }

        // --- scores: S(16x64) = Q(16x64) @ K^T, four 16-col tiles ---
        v8f s[4];
#pragma unroll
        for (int nt = 0; nt < 4; ++nt) {
            s[nt] = zero;
            v16bf kb;
            kb = load_b_frag(&Ks[buf][nt * 16][0], 72, 0,  lane); s[nt] = wmma_bf16(qa[0], kb, s[nt]);
            kb = load_b_frag(&Ks[buf][nt * 16][0], 72, 32, lane); s[nt] = wmma_bf16(qa[1], kb, s[nt]);
        }

        // --- online softmax in exp2 domain (row stats via 16-lane shfl tree) ---
        float alpha[8];
#pragma unroll
        for (int r = 0; r < 8; ++r) {
            float a0 = s[0][r] * cs, a1 = s[1][r] * cs;
            float a2 = s[2][r] * cs, a3 = s[3][r] * cs;
            float mx = fmaxf(fmaxf(a0, a1), fmaxf(a2, a3));
            mx = fmaxf(mx, __shfl_xor(mx, 1));
            mx = fmaxf(mx, __shfl_xor(mx, 2));
            mx = fmaxf(mx, __shfl_xor(mx, 4));
            mx = fmaxf(mx, __shfl_xor(mx, 8));
            float mn = fmaxf(mrow[r], mx);
            alpha[r] = exp2f(mrow[r] - mn);
            a0 = exp2f(a0 - mn); a1 = exp2f(a1 - mn);
            a2 = exp2f(a2 - mn); a3 = exp2f(a3 - mn);
            float rs = (a0 + a1) + (a2 + a3);
            rs += __shfl_xor(rs, 1);
            rs += __shfl_xor(rs, 2);
            rs += __shfl_xor(rs, 4);
            rs += __shfl_xor(rs, 8);
            lrow[r] = lrow[r] * alpha[r] + rs;
            mrow[r] = mn;
            s[0][r] = a0; s[1][r] = a1; s[2][r] = a2; s[3][r] = a3;
        }
#pragma unroll
        for (int f = 0; f < 4; ++f)
#pragma unroll
            for (int r = 0; r < 8; ++r) o[f][r] *= alpha[r];

        // --- P (D-layout) -> per-wave LDS, packed b64 stores (permuted kv) ---
#pragma unroll
        for (int r = 0; r < 8; ++r) {
            ushort4 pk;
            pk.x = f2bf(s[0][r]); pk.y = f2bf(s[1][r]);
            pk.z = f2bf(s[2][r]); pk.w = f2bf(s[3][r]);
            *(ushort4*)&Ps[wave][r + 8 * half][4 * nl] = pk;
        }
        asm volatile("s_wait_dscnt 0x0" ::: "memory");
        v16bf pa[2];
        pa[0] = load_a_frag(&Ps[wave][0][0], 72, 0,  lane);
        pa[1] = load_a_frag(&Ps[wave][0][0], 72, 32, lane);

        // --- O(16x64) += P(16x64) @ V(64x64)  (same kv permutation both sides) ---
#pragma unroll
        for (int f = 0; f < 4; ++f) {
#pragma unroll
            for (int c = 0; c < 2; ++c) {
                v16bf vbf = load_b_frag(&Vt[buf][f * 16][0], 72, c * 32, lane);
                o[f] = wmma_bf16(pa[c], vbf, o[f]);
            }
        }

        // finish staging next tile, then one barrier per iteration
        if (nxt < S) {
            store_v(buf ^ 1);
            if (wave == 0) __builtin_amdgcn_s_wait_tensorcnt(0);
        }
        __syncthreads();
        buf ^= 1;
    }

    // --- normalize (v_rcp + mul, not full div) and store bf16 output ---
#pragma unroll
    for (int r = 0; r < 8; ++r) {
        const float inv = __builtin_amdgcn_rcpf(lrow[r]);
        const int row = q0 + wave * 16 + r + 8 * half;
#pragma unroll
        for (int f = 0; f < 4; ++f)
            Ob[(baseRow + row) * D + hoff + f * 16 + nl] = f2bf(o[f][r] * inv);
    }
}

// ---------------------------------------------------------------------------
// Launch
// ---------------------------------------------------------------------------
extern "C" void kernel_launch(void* const* d_in, const int* in_sizes, int n_in,
                              void* d_out, int out_size, void* d_ws, size_t ws_size,
                              hipStream_t stream)
{
    (void)in_sizes; (void)n_in; (void)out_size; (void)ws_size;
    const float* Q  = (const float*)d_in[0];
    const float* K  = (const float*)d_in[1];
    const float* V  = (const float*)d_in[2];
    const float* Wq = (const float*)d_in[3];
    const float* bq = (const float*)d_in[4];
    const float* Wk = (const float*)d_in[5];
    const float* bk = (const float*)d_in[6];
    const float* Wv = (const float*)d_in[7];
    const float* bv = (const float*)d_in[8];
    const float* Wo = (const float*)d_in[9];
    const float* bo = (const float*)d_in[10];
    float* out = (float*)d_out;

    const int M = 4 * 2048, N = 1024, Kd = 1024;
    unsigned short* qb = (unsigned short*)d_ws;
    unsigned short* kb = qb + (size_t)M * N;
    unsigned short* vb = kb + (size_t)M * N;
    unsigned short* ob = vb + (size_t)M * N;

    dim3 gGemm(N / 128, M / 128);
    gemm_bias_kernel<float, unsigned short><<<gGemm, 256, 0, stream>>>(Q, Wq, bq, qb, M, N, Kd);
    gemm_bias_kernel<float, unsigned short><<<gGemm, 256, 0, stream>>>(K, Wk, bk, kb, M, N, Kd);
    gemm_bias_kernel<float, unsigned short><<<gGemm, 256, 0, stream>>>(V, Wv, bv, vb, M, N, Kd);

    dim3 gAttn(2048 / 128, 16, 4);
    flash_attn_kernel<<<gAttn, 256, 0, stream>>>(qb, kb, vb, ob);

    gemm_bias_kernel<unsigned short, float><<<gGemm, 256, 0, stream>>>(ob, Wo, bo, out, M, N, Kd);
}
